// MoE_Tutel_14396730376784
// MI455X (gfx1250) — matrix operand order repaired and checked by reading, not compiled, not run on hardware
//
#include <hip/hip_runtime.h>
#include <hip/hip_bf16.h>

typedef __bf16 bf16_t;
typedef __attribute__((ext_vector_type(16))) __bf16 v16bf;
typedef __attribute__((ext_vector_type(8)))  __bf16 v8bf;
typedef __attribute__((ext_vector_type(4)))  __bf16 v4bf;
typedef __attribute__((ext_vector_type(8)))  float  v8f;

#define T_TOK 8192
#define DDIM  1024
#define HDIM  2048
#define NEXP  8
#define TM    128
#define TN    64
#define TKC   32
#define LDST  48      // LDS row stride in bf16 elements (96B, 32B-aligned rows)
#define MAX_TILES 256
#define SLOT_CAP  17408   // T*K + E*127 rounded up

__device__ inline void atomicAddF(float* p, float v) {
    __hip_atomic_fetch_add(p, v, __ATOMIC_RELAXED, __HIP_MEMORY_SCOPE_AGENT);
}

// 32-bit LDS offset of a generic pointer known to point into LDS
__device__ inline uint32_t lds_off(const void* p) {
    return (uint32_t)(unsigned long)(__attribute__((address_space(3))) const char*)p;
}
// CDNA5 async DMA: global (16B per lane) -> LDS, tracked by ASYNCcnt
__device__ inline void async_load_b128(uint32_t lds_addr, const void* gptr) {
    asm volatile("global_load_async_to_lds_b128 %0, %1, off"
                 :: "v"(lds_addr), "v"((unsigned long long)(uintptr_t)gptr)
                 : "memory");
}
__device__ inline void wait_asynccnt0() {
    asm volatile("s_wait_asynccnt 0x0" ::: "memory");
}

// ---------------- gating: 1 wave per token ----------------
__global__ __launch_bounds__(256) void gate_kernel(
    const float* __restrict__ x, const float* __restrict__ wg,
    int* __restrict__ topi, float* __restrict__ topv,
    int* __restrict__ count, float* __restrict__ me_sum, float* __restrict__ ce_cnt)
{
    const int lane = threadIdx.x & 31;
    const int wv   = threadIdx.x >> 5;
    const int t    = blockIdx.x * 8 + wv;
    if (t >= T_TOK) return;

    float acc[NEXP];
#pragma unroll
    for (int e = 0; e < NEXP; ++e) acc[e] = 0.f;

    const float* xr = x + (size_t)t * DDIM;
#pragma unroll 4
    for (int i = 0; i < DDIM / 32; ++i) {
        const int d  = i * 32 + lane;
        const float xi = xr[d];
        const float4 w0 = *(const float4*)(wg + d * NEXP);
        const float4 w1 = *(const float4*)(wg + d * NEXP + 4);
        acc[0] += xi * w0.x; acc[1] += xi * w0.y; acc[2] += xi * w0.z; acc[3] += xi * w0.w;
        acc[4] += xi * w1.x; acc[5] += xi * w1.y; acc[6] += xi * w1.z; acc[7] += xi * w1.w;
    }
#pragma unroll
    for (int off = 16; off > 0; off >>= 1) {
#pragma unroll
        for (int e = 0; e < NEXP; ++e) acc[e] += __shfl_xor(acc[e], off, 32);
    }
    if (lane == 0) {
        float mx = acc[0];
#pragma unroll
        for (int e = 1; e < NEXP; ++e) mx = fmaxf(mx, acc[e]);
        float p[NEXP], s = 0.f;
#pragma unroll
        for (int e = 0; e < NEXP; ++e) { p[e] = __expf(acc[e] - mx); s += p[e]; }
        const float inv = 1.f / s;
#pragma unroll
        for (int e = 0; e < NEXP; ++e) p[e] *= inv;
        int i1 = 0;
#pragma unroll
        for (int e = 1; e < NEXP; ++e) if (p[e] > p[i1]) i1 = e;
        int i2 = (i1 == 0) ? 1 : 0;
#pragma unroll
        for (int e = 0; e < NEXP; ++e) if (e != i1 && p[e] > p[i2]) i2 = e;
        topi[t * 2 + 0] = i1; topi[t * 2 + 1] = i2;
        topv[t * 2 + 0] = p[i1]; topv[t * 2 + 1] = p[i2];
        atomicAdd(&count[i1], 1);
        atomicAdd(&count[i2], 1);
        atomicAddF(&ce_cnt[i1], 1.0f);
#pragma unroll
        for (int e = 0; e < NEXP; ++e) atomicAddF(&me_sum[e], p[e]);
    }
}

// ---------------- scan: aligned bases, tile map, l_aux ----------------
__global__ void scan_kernel(const int* __restrict__ count,
                            int* __restrict__ base, int* __restrict__ pcount,
                            int* __restrict__ tile_e, int* __restrict__ tile_m,
                            int* __restrict__ n_tiles,
                            const float* __restrict__ me_sum,
                            const float* __restrict__ ce_cnt,
                            float* __restrict__ laux_out)
{
    if (threadIdx.x != 0 || blockIdx.x != 0) return;
    int off = 0, tiles = 0;
    for (int e = 0; e < NEXP; ++e) {
        base[e] = off;
        const int pc = (count[e] + TM - 1) & ~(TM - 1);
        pcount[e] = pc;
        const int nt = pc / TM;
        for (int i = 0; i < nt && tiles + i < MAX_TILES; ++i) {
            tile_e[tiles + i] = e;
            tile_m[tiles + i] = off + i * TM;
        }
        tiles += nt; off += pc;
    }
    *n_tiles = tiles;
    float l = 0.f;
    for (int e = 0; e < NEXP; ++e)
        l += (me_sum[e] / (float)T_TOK) * (ce_cnt[e] / (float)T_TOK);
    *laux_out = l * (float)NEXP;
}

// ---------------- scatter tokens to slots ----------------
__global__ __launch_bounds__(256) void scatter_kernel(
    const int* __restrict__ topi, const float* __restrict__ topv,
    const int* __restrict__ base, int* __restrict__ cursor,
    int* __restrict__ rows, float* __restrict__ cw)
{
    const int t = blockIdx.x * 256 + threadIdx.x;
    if (t >= T_TOK) return;
#pragma unroll
    for (int k = 0; k < 2; ++k) {
        const int e = topi[t * 2 + k];
        const int pos = atomicAdd(&cursor[e], 1);
        const int slot = base[e] + pos;
        rows[slot] = t;
        cw[slot] = topv[t * 2 + k];
    }
}

// ---------------- grouped GEMM, 128x64 tile, bf16 WMMA, double-buffered ----
// MODE 0: h[slot,:H] = relu( gather(x)[slot,:D] @ w1[e] + b1[e] )   (K=D, N=H)
// MODE 1: out[tok,:D] += cw[slot] * ( h[slot,:H] @ w2[e] + b2[e] )  (K=H, N=D)
template <int MODE>
__global__ __launch_bounds__(256) void ffn_kernel(
    const float*  __restrict__ x,
    const bf16_t* __restrict__ hsrc,
    const float*  __restrict__ w,
    const float*  __restrict__ bias,
    bf16_t* __restrict__ hdst,
    float*  __restrict__ out,
    const int* __restrict__ tile_e, const int* __restrict__ tile_m,
    const int* __restrict__ n_tiles_p,
    const int* __restrict__ rows, const float* __restrict__ cw,
    int K, int N)
{
    __shared__ __align__(32) bf16_t As[2][TM][LDST];
    __shared__ __align__(32) bf16_t Bs[2][TN][LDST];
    __shared__ int   s_tok[TM];
    __shared__ float s_cw[TM];

    const int bx = blockIdx.x;
    if (bx >= *n_tiles_p) return;
    const int e      = tile_e[bx];
    const int m_base = tile_m[bx];
    const int n_base = blockIdx.y * TN;
    const int tid  = threadIdx.x;
    const int lane = tid & 31;
    const int wv   = tid >> 5;

    if (tid < TM) {
        s_tok[tid] = rows[m_base + tid];
        s_cw[tid]  = cw[m_base + tid];
    }
    __syncthreads();

    const float* wbase = w + (size_t)e * K * N;

    // ---- per-thread staging coordinates (loop-invariant) ----
    const int bK0 = tid >> 4;            // B: k row (0..15), +16 for it=1
    const int bN0 = (tid * 4) & 63;      // B: 4 consecutive n
    const float* bptr0 = wbase + (size_t)bK0 * N + n_base + bN0;

    // MODE0 A gather: 4 chunks of float4
    const int aRow0 = tid >> 3;          // +it*32
    const int aCol  = (tid * 4) & 31;
    const float* aptr[4];
    float amask[4];
    // MODE1 A async: 2 chunks of 16B
    const int hRow0 = tid >> 2;          // +it*64
    const int hCol  = (tid * 8) & 31;
    const bf16_t* hptr[2];
    if constexpr (MODE == 0) {
#pragma unroll
        for (int it = 0; it < 4; ++it) {
            const int tok = s_tok[aRow0 + it * 32];
            amask[it] = (tok >= 0) ? 1.f : 0.f;
            aptr[it] = x + (size_t)(tok < 0 ? 0 : tok) * K + aCol;
        }
    } else {
#pragma unroll
        for (int it = 0; it < 2; ++it)
            hptr[it] = hsrc + (size_t)(m_base + hRow0 + it * 64) * K + hCol;
    }

    // ---- prologue: stage tile kb=0 into buffer 0 ----
    {
#pragma unroll
        for (int it = 0; it < 2; ++it) {
            const float4 v = *(const float4*)(bptr0 + (size_t)(it * 16) * N);
            const int k = bK0 + it * 16;
            Bs[0][bN0 + 0][k] = (bf16_t)v.x;
            Bs[0][bN0 + 1][k] = (bf16_t)v.y;
            Bs[0][bN0 + 2][k] = (bf16_t)v.z;
            Bs[0][bN0 + 3][k] = (bf16_t)v.w;
        }
        if constexpr (MODE == 0) {
#pragma unroll
            for (int it = 0; it < 4; ++it) {
                const float4 v = *(const float4*)(aptr[it]);
                const float m = amask[it];
                v4bf p = { (bf16_t)(v.x * m), (bf16_t)(v.y * m),
                           (bf16_t)(v.z * m), (bf16_t)(v.w * m) };
                *(v4bf*)&As[0][aRow0 + it * 32][aCol] = p;
            }
        } else {
#pragma unroll
            for (int it = 0; it < 2; ++it)
                async_load_b128(lds_off(&As[0][hRow0 + it * 64][hCol]), hptr[it]);
            wait_asynccnt0();
        }
    }
    __syncthreads();

    v8f acc[4] = {};
    const int mrow = wv * 16 + (lane & 15);
    const int ka   = (lane & 16) ? 8 : 0;
    const int kb0  = (lane & 16) ? 16 : 0;

    int buf = 0;
    for (int kb = 0; kb < K; kb += TKC) {
        const int kn = kb + TKC;
        const bool nxt = kn < K;
        // ---- issue next tile's global loads (regs / async into buf^1) ----
        float4 bpr[2];
        float4 apr[4];
        if (nxt) {
#pragma unroll
            for (int it = 0; it < 2; ++it)
                bpr[it] = *(const float4*)(bptr0 + (size_t)(kn + it * 16) * N);
            if constexpr (MODE == 0) {
#pragma unroll
                for (int it = 0; it < 4; ++it)
                    apr[it] = *(const float4*)(aptr[it] + kn);
            } else {
#pragma unroll
                for (int it = 0; it < 2; ++it)
                    async_load_b128(lds_off(&As[buf ^ 1][hRow0 + it * 64][hCol]),
                                    hptr[it] + kn);
            }
        }
        // ---- compute current buffer: 1 A frag, 4 B frags, 4 WMMAs ----
        const v8bf alo = *(const v8bf*)&As[buf][mrow][kb0];
        const v8bf ahi = *(const v8bf*)&As[buf][mrow][kb0 + 8];
        const v16bf afrag = __builtin_shufflevector(alo, ahi,
            0, 1, 2, 3, 4, 5, 6, 7, 8, 9, 10, 11, 12, 13, 14, 15);
        v16bf bfrag[4];
#pragma unroll
        for (int sub = 0; sub < 4; ++sub)
            bfrag[sub] = *(const v16bf*)&Bs[buf][sub * 16 + (lane & 15)][kb0];
#pragma unroll
        for (int sub = 0; sub < 4; ++sub)
            acc[sub] = __builtin_amdgcn_wmma_f32_16x16x32_bf16(
                false, afrag, false, bfrag[sub], (short)0, acc[sub], false, false);
        // ---- convert + store prefetched data into buf^1 ----
        if (nxt) {
#pragma unroll
            for (int it = 0; it < 2; ++it) {
                const int k = bK0 + it * 16;
                Bs[buf ^ 1][bN0 + 0][k] = (bf16_t)bpr[it].x;
                Bs[buf ^ 1][bN0 + 1][k] = (bf16_t)bpr[it].y;
                Bs[buf ^ 1][bN0 + 2][k] = (bf16_t)bpr[it].z;
                Bs[buf ^ 1][bN0 + 3][k] = (bf16_t)bpr[it].w;
            }
            if constexpr (MODE == 0) {
#pragma unroll
                for (int it = 0; it < 4; ++it) {
                    const float m = amask[it];
                    v4bf p = { (bf16_t)(apr[it].x * m), (bf16_t)(apr[it].y * m),
                               (bf16_t)(apr[it].z * m), (bf16_t)(apr[it].w * m) };
                    *(v4bf*)&As[buf ^ 1][aRow0 + it * 32][aCol] = p;
                }
            } else {
                wait_asynccnt0();
            }
        }
        __syncthreads();
        buf ^= 1;
    }

    // ---- epilogue ----
    const int col0  = n_base + (lane & 15);
    const int rbase = wv * 16 + ((lane & 16) ? 8 : 0);
    if constexpr (MODE == 0) {
#pragma unroll
        for (int sub = 0; sub < 4; ++sub) {
            const float bv = bias[(size_t)e * N + col0 + sub * 16];
#pragma unroll
            for (int r = 0; r < 8; ++r) {
                float vsum = acc[sub][r] + bv;
                vsum = vsum > 0.f ? vsum : 0.f;
                const int slot = m_base + rbase + r;
                hdst[(size_t)slot * N + col0 + sub * 16] = (bf16_t)vsum;
            }
        }
    } else {
#pragma unroll
        for (int r = 0; r < 8; ++r) {
            const int lr = rbase + r;
            const int tok = s_tok[lr];
            if (tok < 0) continue;
            const float cwv = s_cw[lr];
#pragma unroll
            for (int sub = 0; sub < 4; ++sub) {
                const float bv = bias[(size_t)e * N + col0 + sub * 16];
                const float vsum = (acc[sub][r] + bv) * cwv;
                atomicAddF(out + (size_t)tok * N + col0 + sub * 16, vsum);
            }
        }
    }
}

extern "C" void kernel_launch(void* const* d_in, const int* in_sizes, int n_in,
                              void* d_out, int out_size, void* d_ws, size_t ws_size,
                              hipStream_t stream) {
    (void)in_sizes; (void)n_in; (void)out_size; (void)ws_size;
    const float* x  = (const float*)d_in[0];
    const float* wg = (const float*)d_in[1];
    const float* w1 = (const float*)d_in[2];
    const float* b1 = (const float*)d_in[3];
    const float* w2 = (const float*)d_in[4];
    const float* b2 = (const float*)d_in[5];
    float* out = (float*)d_out;

    char* ws = (char*)d_ws;
    int*   count  = (int*)(ws + 0);
    int*   cursor = (int*)(ws + 32);
    int*   base   = (int*)(ws + 64);
    int*   pcount = (int*)(ws + 96);
    int*   tile_e = (int*)(ws + 128);
    int*   tile_m = (int*)(ws + 1152);
    int*   ntile  = (int*)(ws + 2176);
    float* me_sum = (float*)(ws + 2208);
    float* ce_cnt = (float*)(ws + 2240);
    int*   topi   = (int*)(ws + 2304);
    float* topv   = (float*)(ws + 2304 + 65536);
    int*   rows   = (int*)(ws + 2304 + 131072);
    float* cw     = (float*)(ws + 2304 + 131072 + SLOT_CAP * 4);
    bf16_t* h     = (bf16_t*)(ws + 2304 + 131072 + 2 * SLOT_CAP * 4);

    // zero counters/stats; rows = -1; cw = 0; out = 0
    hipMemsetAsync(ws, 0, 2304, stream);
    hipMemsetAsync(rows, 0xFF, SLOT_CAP * 4, stream);
    hipMemsetAsync(cw, 0, SLOT_CAP * 4, stream);
    hipMemsetAsync(out, 0, ((size_t)T_TOK * DDIM + 1) * sizeof(float), stream);

    gate_kernel<<<T_TOK / 8, 256, 0, stream>>>(x, wg, topi, topv, count, me_sum, ce_cnt);
    scan_kernel<<<1, 32, 0, stream>>>(count, base, pcount, tile_e, tile_m, ntile,
                                      me_sum, ce_cnt, out + (size_t)T_TOK * DDIM);
    scatter_kernel<<<T_TOK / 256, 256, 0, stream>>>(topi, topv, base, cursor, rows, cw);

    const int max_tiles = T_TOK * 2 / TM + NEXP;  // 136 worst case
    ffn_kernel<0><<<dim3(max_tiles, HDIM / TN), 256, 0, stream>>>(
        x, nullptr, w1, b1, h, nullptr, tile_e, tile_m, ntile, rows, cw, DDIM, HDIM);
    ffn_kernel<1><<<dim3(max_tiles, DDIM / TN), 256, 0, stream>>>(
        nullptr, h, w2, b2, nullptr, out, tile_e, tile_m, ntile, rows, cw, HDIM, DDIM);
}
